// SoftAggONNX_v1_77730318123554
// MI455X (gfx1250) — matrix operand order repaired and checked by reading, not compiled, not run on hardware
//
#include <hip/hip_runtime.h>
#include <stdint.h>

typedef __attribute__((ext_vector_type(16))) __bf16 v16bf;
typedef __attribute__((ext_vector_type(8)))  float  v8f;

#define D_DIM 384
#define LDS_STRIDE 388   // f32 per LDS row: 384 + 4 pad (bank spread)
#define N_SEQ 16384

// ---------------- small helpers ----------------

__device__ __forceinline__ unsigned short f2bf(float f) {
  unsigned int u = __float_as_uint(f);
  u += 0x7FFFu + ((u >> 16) & 1u);     // round-to-nearest-even
  return (unsigned short)(u >> 16);
}

// CDNA5 async global->LDS DMA (GVS mode: saddr64 + vaddr32)
__device__ __forceinline__ void async_ld16(uint32_t lds_off, uint64_t gbase, uint32_t goff) {
  asm volatile("global_load_async_to_lds_b128 %0, %1, %2"
               :: "v"(lds_off), "v"(goff), "s"(gbase) : "memory");
}
__device__ __forceinline__ void wait_async0() {
  asm volatile("s_wait_asynccnt 0" ::: "memory");
}

// Pull a 32-row x 384-col f32 tile into LDS (padded) with the async-copy path.
__device__ __forceinline__ void load_tile_async(const float* __restrict__ gsrc,
                                                int row0, float* xs) {
  uint32_t lds_base = (uint32_t)(uintptr_t)xs;   // low 32 bits of generic ptr = LDS offset
  uint64_t gb = (uint64_t)(uintptr_t)gsrc;
  int tid = threadIdx.x;
  #pragma unroll
  for (int i = 0; i < 12; ++i) {                 // 3072 16B-chunks / 256 threads
    int c = tid + i * 256;
    int r = c / 96;
    int q = c - r * 96;
    uint32_t goff = (uint32_t)(((uint32_t)(row0 + r) * D_DIM + q * 4) * 4);
    uint32_t loff = lds_base + (uint32_t)((r * LDS_STRIDE + q * 4) * 4);
    async_ld16(loff, gb, goff);
  }
  wait_async0();
  __syncthreads();
}

// A-fragment (16x32 bf16): contiguous layout — lane(hi=0): K0-15; lane(hi=1): K16-31
__device__ __forceinline__ v16bf load_a_frag(const float* arow, int k0, int hi) {
  const float* p = arow + k0 + hi * 16;
  const float4 q0 = *(const float4*)(p);
  const float4 q1 = *(const float4*)(p + 4);
  const float4 q2 = *(const float4*)(p + 8);
  const float4 q3 = *(const float4*)(p + 12);
  v16bf a;
  a[0]=(__bf16)q0.x; a[1]=(__bf16)q0.y; a[2]=(__bf16)q0.z;  a[3]=(__bf16)q0.w;
  a[4]=(__bf16)q1.x; a[5]=(__bf16)q1.y; a[6]=(__bf16)q1.z;  a[7]=(__bf16)q1.w;
  a[8]=(__bf16)q2.x; a[9]=(__bf16)q2.y; a[10]=(__bf16)q2.z; a[11]=(__bf16)q2.w;
  a[12]=(__bf16)q3.x; a[13]=(__bf16)q3.y; a[14]=(__bf16)q3.z; a[15]=(__bf16)q3.w;
  return a;
}

// ---------------- kernels ----------------

__global__ __launch_bounds__(256) void cvt_bf16_k(const float* __restrict__ src,
                                                  unsigned short* __restrict__ dst) {
  int i = blockIdx.x * 256 + threadIdx.x;        // grid sized exactly
  dst[i] = f2bf(src[i]);
}

__global__ __launch_bounds__(256) void zero4_k(float4* __restrict__ p) {
  p[blockIdx.x * 256 + threadIdx.x] = float4{0.f, 0.f, 0.f, 0.f};
}

// Fused: logits=x@gw^T+gb, fx=x@fw^T+fb, e=exp(logits); atomically segment-sum e and fx*e.
__global__ __launch_bounds__(256) void gemm_gf_k(
    const float* __restrict__ x, const int* __restrict__ ii,
    const unsigned short* __restrict__ gw, const float* __restrict__ gb,
    const unsigned short* __restrict__ fw, const float* __restrict__ fb,
    float* __restrict__ acc_e, float* __restrict__ acc_fe) {
  __shared__ float xs[32 * LDS_STRIDE];
  const int row0 = blockIdx.x * 32;
  load_tile_async(x, row0, xs);

  const int tid = threadIdx.x;
  const int lane = tid & 31, w = tid >> 5;
  const int wm = w & 1, wn = w >> 1;             // 2 row-groups x 4 col-groups of waves
  const int m = lane & 15, hi = lane >> 4;

  v8f accG[6], accF[6];
  #pragma unroll
  for (int t = 0; t < 6; ++t) { accG[t] = 0.f; accF[t] = 0.f; }

  const float* arow = xs + (wm * 16 + m) * LDS_STRIDE;
  const int nb = wn * 96 + m;                    // per-lane B column index base
  // Per-lane B base pointers; all tile/k offsets become compile-time immediates.
  const __bf16* gB = (const __bf16*)gw + (size_t)nb * D_DIM + hi * 16;
  const __bf16* fB = (const __bf16*)fw + (size_t)nb * D_DIM + hi * 16;

  #pragma unroll
  for (int k0 = 0; k0 < D_DIM; k0 += 32) {
    v16bf a = load_a_frag(arow, k0, hi);
    #pragma unroll
    for (int t = 0; t < 6; ++t) {
      v16bf bg = *(const v16bf*)(gB + t * 16 * D_DIM + k0);
      accG[t] = __builtin_amdgcn_wmma_f32_16x16x32_bf16(false, a, false, bg,
                                                        (short)0, accG[t], false, false);
      v16bf bf = *(const v16bf*)(fB + t * 16 * D_DIM + k0);
      accF[t] = __builtin_amdgcn_wmma_f32_16x16x32_bf16(false, a, false, bf,
                                                        (short)0, accF[t], false, false);
    }
  }

  float gbv[6], fbv[6];
  #pragma unroll
  for (int t = 0; t < 6; ++t) {
    int col = wn * 96 + t * 16 + m;
    gbv[t] = gb[col]; fbv[t] = fb[col];
  }

  #pragma unroll
  for (int r = 0; r < 8; ++r) {
    int row = row0 + wm * 16 + hi * 8 + r;       // C layout: VGPR r -> M = hi*8 + r
    int b = row >> 14;                           // row / 16384
    int n = row & (N_SEQ - 1);
    int g = ii[n];
    size_t base = ((size_t)(b * 512 + g)) * D_DIM;
    #pragma unroll
    for (int t = 0; t < 6; ++t) {
      int col = wn * 96 + t * 16 + m;
      float e  = __expf(accG[t][r] + gbv[t]);
      float fe = (accF[t][r] + fbv[t]) * e;
      unsafeAtomicAdd(&acc_e[base + col], e);
      unsafeAtomicAdd(&acc_fe[base + col], fe);
    }
  }
}

__global__ __launch_bounds__(256) void yin_k(const float* __restrict__ acc_fe,
                                             const float* __restrict__ acc_e,
                                             float* __restrict__ yin) {
  int i = blockIdx.x * 256 + threadIdx.x;        // grid sized exactly (786432)
  yin[i] = acc_fe[i] / (acc_e[i] + 1e-6f);
}

// z = yin @ hw^T + hb over 2048 rows
__global__ __launch_bounds__(256) void gemm_h_k(
    const float* __restrict__ yin, const unsigned short* __restrict__ hw,
    const float* __restrict__ hb, float* __restrict__ z) {
  __shared__ float xs[32 * LDS_STRIDE];
  const int row0 = blockIdx.x * 32;
  load_tile_async(yin, row0, xs);

  const int tid = threadIdx.x;
  const int lane = tid & 31, w = tid >> 5;
  const int wm = w & 1, wn = w >> 1;
  const int m = lane & 15, hi = lane >> 4;

  v8f acc[6];
  #pragma unroll
  for (int t = 0; t < 6; ++t) acc[t] = 0.f;

  const float* arow = xs + (wm * 16 + m) * LDS_STRIDE;
  const int nb = wn * 96 + m;
  const __bf16* hB = (const __bf16*)hw + (size_t)nb * D_DIM + hi * 16;

  #pragma unroll
  for (int k0 = 0; k0 < D_DIM; k0 += 32) {
    v16bf a = load_a_frag(arow, k0, hi);
    #pragma unroll
    for (int t = 0; t < 6; ++t) {
      v16bf bh = *(const v16bf*)(hB + t * 16 * D_DIM + k0);
      acc[t] = __builtin_amdgcn_wmma_f32_16x16x32_bf16(false, a, false, bh,
                                                       (short)0, acc[t], false, false);
    }
  }

  float hbv[6];
  #pragma unroll
  for (int t = 0; t < 6; ++t) hbv[t] = hb[wn * 96 + t * 16 + m];

  #pragma unroll
  for (int r = 0; r < 8; ++r) {
    int row = row0 + wm * 16 + hi * 8 + r;
    #pragma unroll
    for (int t = 0; t < 6; ++t) {
      int col = wn * 96 + t * 16 + m;
      z[(size_t)row * D_DIM + col] = acc[t][r] + hbv[t];
    }
  }
}

// out[b,n,:] = z[b*512 + ii[n], :] as float4
__global__ __launch_bounds__(256) void gather_k(const float* __restrict__ z,
                                                const int* __restrict__ ii,
                                                float* __restrict__ out) {
  int idx = blockIdx.x * 256 + threadIdx.x;      // float4 index; grid sized exactly
  int row = idx / 96;
  int q = idx - row * 96;
  int b = row >> 14;
  int n = row & (N_SEQ - 1);
  int g = ii[n];
  const float4* src = (const float4*)(z + ((size_t)(b * 512 + g)) * D_DIM) + q;
  ((float4*)out)[idx] = *src;
}

// ---------------- launcher ----------------

extern "C" void kernel_launch(void* const* d_in, const int* in_sizes, int n_in,
                              void* d_out, int out_size, void* d_ws, size_t ws_size,
                              hipStream_t stream) {
  (void)in_sizes; (void)n_in; (void)out_size; (void)ws_size;
  const float* x   = (const float*)d_in[0];
  const int*   ii  = (const int*)d_in[1];
  const float* f_w = (const float*)d_in[2];
  const float* f_b = (const float*)d_in[3];
  const float* g_w = (const float*)d_in[4];
  const float* g_b = (const float*)d_in[5];
  const float* h_w = (const float*)d_in[6];
  const float* h_b = (const float*)d_in[7];
  float* out = (float*)d_out;

  char* ws = (char*)d_ws;
  unsigned short* gw_bf = (unsigned short*)(ws);
  unsigned short* fw_bf = (unsigned short*)(ws + 294912);
  unsigned short* hw_bf = (unsigned short*)(ws + 589824);
  float* acc_e  = (float*)(ws + 884736);
  float* acc_fe = (float*)(ws + 884736 + 3145728);
  float* yin    = (float*)(ws + 884736 + 2 * 3145728);
  float* z      = (float*)(ws + 884736 + 3 * 3145728);   // total ~13.5 MB

  cvt_bf16_k<<<576, 256, 0, stream>>>(g_w, gw_bf);
  cvt_bf16_k<<<576, 256, 0, stream>>>(f_w, fw_bf);
  cvt_bf16_k<<<576, 256, 0, stream>>>(h_w, hw_bf);
  zero4_k<<<1536, 256, 0, stream>>>((float4*)acc_e);      // zeros acc_e + acc_fe (contiguous)
  gemm_gf_k<<<2048, 256, 0, stream>>>(x, ii, gw_bf, g_b, fw_bf, f_b, acc_e, acc_fe);
  yin_k<<<3072, 256, 0, stream>>>(acc_fe, acc_e, yin);
  gemm_h_k<<<64, 256, 0, stream>>>(yin, hw_bf, h_b, z);
  gather_k<<<24576, 256, 0, stream>>>(z, ii, out);
}
